// OneHopExpert_58342835749141
// MI455X (gfx1250) — matrix operand order repaired and checked by reading, not compiled, not run on hardware
//
#include <hip/hip_runtime.h>
#include <hip/hip_bf16.h>

// ---------------------------------------------------------------------------
// CDNA5 (gfx1250) 2-layer edge-conditioned GCN.
// GEMMs: v_wmma_f32_16x16x32_bf16 (fp32 accum), LDS-staged.
// A fragments are register-resident per wave (fixed M-tile per wave),
// B streams from LDS-transposed weights as ds_load_b128 pairs.
// ---------------------------------------------------------------------------

typedef __attribute__((ext_vector_type(16))) __bf16        v16bf;
typedef __attribute__((ext_vector_type(4)))  __bf16        v4bf;
typedef __attribute__((ext_vector_type(8)))  float         v8f;
typedef __attribute__((ext_vector_type(4)))  unsigned int  v4u;

#define LN_EPS 1e-5f

__device__ __forceinline__ __bf16 f2bf(float f) {
    unsigned int u = __float_as_uint(f);
    unsigned int r = (u + 0x7FFFu + ((u >> 16) & 1u)) >> 16;   // RNE
    unsigned short s = (unsigned short)r;
    __bf16 b;
    __builtin_memcpy(&b, &s, 2);
    return b;
}

union frag_cast { v16bf v; v4u q[2]; };

// A-fragment: 16x32 bf16 tile, LDS row-major, stride in elements (mult of 8).
// ISA 7.12.2: lane half selects K base {0,8}; per-lane data = p[0..7], p[16..23]
// -> two contiguous 16B runs -> two ds_load_b128.
__device__ __forceinline__ v16bf load_frag_a(const __bf16* lds, int row0,
                                             int stride, int k0, int lane) {
    int m  = lane & 15;
    int kb = (lane >> 4) << 3;                 // 0 or 8
    const __bf16* p = lds + (size_t)(row0 + m) * stride + k0 + kb;
    frag_cast f;
    f.q[0] = *(const v4u*)(p);
    f.q[1] = *(const v4u*)(p + 16);
    return f.v;
}

// B-fragment from LDS-transposed weights wT[n][k] (stride mult of 8).
// Lanes 0-15: N=n0+lane, K=k0+{0..7,16..23} ; lanes 16-31: K=k0+{8..15,24..31}.
// Per-lane data = p[0..7], p[16..23] (two 16B runs) -> two ds_load_b128.
__device__ __forceinline__ v16bf load_frag_bt(const __bf16* wt, int n0,
                                              int stride, int k0, int lane) {
    int n  = n0 + (lane & 15);
    int kb = (lane >> 4) << 3;                 // 0 or 8
    const __bf16* p = wt + (size_t)n * stride + k0 + kb;
    frag_cast f;
    f.q[0] = *(const v4u*)(p);
    f.q[1] = *(const v4u*)(p + 16);
    return f.v;
}

__device__ __forceinline__ v8f wmma_bf16(v16bf a, v16bf b, v8f c) {
    return __builtin_amdgcn_wmma_f32_16x16x32_bf16(
        false, a, false, b, (short)0, c, false, false);
}

__device__ __forceinline__ float wave_sum32(float v) {
#pragma unroll
    for (int off = 16; off > 0; off >>= 1) v += __shfl_xor(v, off, 32);
    return v;
}

__device__ __forceinline__ float gelu_exact(float v) {
    return 0.5f * v * (1.0f + erff(v * 0.7071067811865476f));
}

__device__ __forceinline__ v4bf pack4(float4 v) {
    v4bf b;
    b[0] = f2bf(v.x); b[1] = f2bf(v.y); b[2] = f2bf(v.z); b[3] = f2bf(v.w);
    return b;
}

// ---------------------------------------------------------------------------
__global__ void zero_kernel(float* __restrict__ p, long long n) {
    long long i = (long long)blockIdx.x * blockDim.x + threadIdx.x;
    long long s = (long long)gridDim.x * blockDim.x;
    for (; i < n; i += s) p[i] = 0.0f;
}

__global__ void dinv_kernel(const float* __restrict__ deg,
                            float* __restrict__ dinv, int n) {
    int i = blockIdx.x * blockDim.x + threadIdx.x;
    if (i < n) dinv[i] = rsqrtf(deg[i] + 1.0f);
}

// ---------------------------------------------------------------------------
// Edge init: ef = relu([x_r||x_c] @ ei_w1 + b1) @ ei_w2 + b2 ; ew = sigmoid(..)
// 128 edges per block (8 waves). LDS strides padded (+8 elems) vs bank aliasing.
//   s_in  [128][264] bf16  @ 0       (67584)
//   s_w1t [ 64][264] bf16  @ 67584   (33792)   (= ei_w1^T)
//   s_h   [128][ 72] bf16  @ 101376  (18432)
//   s_w2t [ 64][ 72] bf16  @ 119808  ( 9216)   (= ei_w2^T)
//   s_ef  [128][ 68] f32   @ 129024  (34816)
//   s_row [128] int        @ 163840, s_col @ 164352 ; total 164864 B
// ---------------------------------------------------------------------------
__global__ void edge_init_kernel(
    const float* __restrict__ x, const int* __restrict__ ei,
    const float* __restrict__ w1, const float* __restrict__ b1,
    const float* __restrict__ w2, const float* __restrict__ b2,
    const float* __restrict__ eww, const float* __restrict__ ewb,
    float* __restrict__ ef, float* __restrict__ ew,
    float* __restrict__ deg, int nE)
{
    extern __shared__ char smem[];
    __bf16* s_in  = (__bf16*)(smem);
    __bf16* s_w1t = (__bf16*)(smem + 67584);
    __bf16* s_h   = (__bf16*)(smem + 101376);
    __bf16* s_w2t = (__bf16*)(smem + 119808);
    float*  s_ef  = (float*)(smem + 129024);
    int*    s_row = (int*)(smem + 163840);
    int*    s_col = (int*)(smem + 164352);

    const int tid = threadIdx.x;
    const int e0  = blockIdx.x * 128;

    for (int i = tid; i < 128; i += 256) {
        int e = e0 + i; if (e > nE - 1) e = nE - 1;
        s_row[i] = ei[e];
        s_col[i] = ei[nE + e];
    }
    __syncthreads();

    // gather [x[row] || x[col]] -> bf16 LDS (float4 loads, v4bf stores)
    for (int idx = tid; idx < 128 * 64; idx += 256) {
        int le = idx >> 6, c4 = (idx & 63) << 2;   // c4 in 0..252
        const float* src = (c4 < 128)
            ? (x + (size_t)s_row[le] * 128 + c4)
            : (x + (size_t)s_col[le] * 128 + (c4 - 128));
        *(v4bf*)(s_in + (size_t)le * 264 + c4) = pack4(*(const float4*)src);
    }
    // weights transposed: s_w1t[n][k] = w1[k][n]
    for (int idx = tid; idx < 256 * 64; idx += 256) {
        int k = idx >> 6, n = idx & 63;
        s_w1t[(size_t)n * 264 + k] = f2bf(w1[idx]);
    }
    for (int idx = tid; idx < 64 * 64; idx += 256) {
        int k = idx >> 6, n = idx & 63;
        s_w2t[(size_t)n * 72 + k] = f2bf(w2[idx]);
    }
    __syncthreads();

    const int wv = tid >> 5, lane = tid & 31;

    // GEMM1: [128,256] @ [256,64] -> relu -> s_h (bf16)
    // Each wave owns M-tile mt=wv; A fragments register-resident across nt.
    {
        const int mt = wv;
        v16bf afr[8];
#pragma unroll
        for (int kk = 0; kk < 8; ++kk)
            afr[kk] = load_frag_a(s_in, mt * 16, 264, kk * 32, lane);
        for (int nt = 0; nt < 4; ++nt) {
            v8f acc = {};
#pragma unroll
            for (int kk = 0; kk < 8; ++kk)
                acc = wmma_bf16(afr[kk],
                                load_frag_bt(s_w1t, nt * 16, 264, kk * 32, lane),
                                acc);
            int ccol  = nt * 16 + (lane & 15);
            float bi  = b1[ccol];
            int rbase = mt * 16 + ((lane >> 4) << 3);
#pragma unroll
            for (int r = 0; r < 8; ++r) {
                float v = acc[r] + bi;
                v = v > 0.0f ? v : 0.0f;
                s_h[(size_t)(rbase + r) * 72 + ccol] = f2bf(v);
            }
        }
    }
    __syncthreads();

    // GEMM2: [128,64] @ [64,64] + b2 -> s_ef (f32) + global ef
    {
        const int mt = wv;
        v16bf afr[2];
#pragma unroll
        for (int kk = 0; kk < 2; ++kk)
            afr[kk] = load_frag_a(s_h, mt * 16, 72, kk * 32, lane);
        for (int nt = 0; nt < 4; ++nt) {
            v8f acc = {};
#pragma unroll
            for (int kk = 0; kk < 2; ++kk)
                acc = wmma_bf16(afr[kk],
                                load_frag_bt(s_w2t, nt * 16, 72, kk * 32, lane),
                                acc);
            int ccol  = nt * 16 + (lane & 15);
            float bi  = b2[ccol];
            int rbase = mt * 16 + ((lane >> 4) << 3);
#pragma unroll
            for (int r = 0; r < 8; ++r) {
                float v = acc[r] + bi;
                int le = rbase + r;
                s_ef[(size_t)le * 68 + ccol] = v;
                int e = e0 + le;
                if (e < nE) ef[(size_t)e * 64 + ccol] = v;
            }
        }
    }
    __syncthreads();

    // edge weight + degree accumulation (wave-parallel dot over 64 feats)
    for (int le = wv; le < 128; le += 8) {
        float v0 = s_ef[(size_t)le * 68 + lane];
        float v1 = s_ef[(size_t)le * 68 + 32 + lane];
        float dot = wave_sum32(v0 * eww[lane] + v1 * eww[32 + lane]) + ewb[0];
        float w = 1.0f / (1.0f + expf(-dot));
        int e = e0 + le;
        if (e < nE && lane == 0) {
            ew[e] = w;
            atomicAdd(&deg[s_col[le]], w);
        }
    }
}

// ---------------------------------------------------------------------------
// xw = A[nrows,128] @ W[128,128].  s_a [128][136] @0, s_wt [128][136] @34816.
// ---------------------------------------------------------------------------
__global__ void gemm_node_kernel(const float* __restrict__ A,
                                 const float* __restrict__ W,
                                 float* __restrict__ out, int nrows)
{
    extern __shared__ char smem[];
    __bf16* s_a  = (__bf16*)(smem);
    __bf16* s_wt = (__bf16*)(smem + 34816);

    const int tid = threadIdx.x;
    const int r0  = blockIdx.x * 128;

    for (int idx = tid; idx < 128 * 32; idx += 256) {
        int lr = idx >> 5;
        int r = r0 + lr; if (r > nrows - 1) r = nrows - 1;
        int c4 = (idx & 31) << 2;
        *(v4bf*)(s_a + (size_t)lr * 136 + c4) =
            pack4(*(const float4*)(A + (size_t)r * 128 + c4));
    }
    for (int idx = tid; idx < 128 * 128; idx += 256) {
        int k = idx >> 7, n = idx & 127;
        s_wt[(size_t)n * 136 + k] = f2bf(W[idx]);
    }
    __syncthreads();

    const int wv = tid >> 5, lane = tid & 31;
    {
        const int mt = wv;
        v16bf afr[4];
#pragma unroll
        for (int kk = 0; kk < 4; ++kk)
            afr[kk] = load_frag_a(s_a, mt * 16, 136, kk * 32, lane);
        for (int nt = 0; nt < 8; ++nt) {
            v8f acc = {};
#pragma unroll
            for (int kk = 0; kk < 4; ++kk)
                acc = wmma_bf16(afr[kk],
                                load_frag_bt(s_wt, nt * 16, 136, kk * 32, lane),
                                acc);
            int ccol  = nt * 16 + (lane & 15);
            int rbase = mt * 16 + ((lane >> 4) << 3);
#pragma unroll
            for (int r = 0; r < 8; ++r) {
                int rr = r0 + rbase + r;
                if (rr < nrows) out[(size_t)rr * 128 + ccol] = acc[r];
            }
        }
    }
}

// ---------------------------------------------------------------------------
// Scatter: agg[col] += dinv[row]*ew*dinv[col] * xw[row].  One wave per edge.
// ---------------------------------------------------------------------------
__global__ void scatter_kernel(const int* __restrict__ ei,
                               const float* __restrict__ ew,
                               const float* __restrict__ dinv,
                               const float* __restrict__ xw,
                               float* __restrict__ agg, int nE)
{
    long long t = (long long)blockIdx.x * blockDim.x + threadIdx.x;
    int e = (int)(t >> 5);
    if (e >= nE) return;
    int lane = threadIdx.x & 31;
    int r = ei[e], c = ei[nE + e];
    float norm = dinv[r] * ew[e] * dinv[c];
    float4 v = ((const float4*)(xw + (size_t)r * 128))[lane];
    float* dst = agg + (size_t)c * 128 + lane * 4;
    atomicAdd(dst + 0, norm * v.x);
    atomicAdd(dst + 1, norm * v.y);
    atomicAdd(dst + 2, norm * v.z);
    atomicAdd(dst + 3, norm * v.w);
}

// ---------------------------------------------------------------------------
// Node finalize: LN(gelu(agg + dinv^2 * xw + bias)).  One wave per node.
// ---------------------------------------------------------------------------
__global__ void node_finalize_kernel(const float* __restrict__ agg,
                                     const float* __restrict__ xw,
                                     const float* __restrict__ dinv,
                                     const float* __restrict__ bias,
                                     const float* __restrict__ g,
                                     const float* __restrict__ bta,
                                     float* __restrict__ out, int nN)
{
    long long t = (long long)blockIdx.x * blockDim.x + threadIdx.x;
    int n = (int)(t >> 5);
    if (n >= nN) return;
    int lane = threadIdx.x & 31;

    float di = dinv[n]; float di2 = di * di;
    float4 a4 = ((const float4*)(agg + (size_t)n * 128))[lane];
    float4 x4 = ((const float4*)(xw  + (size_t)n * 128))[lane];
    int c0 = lane * 4;
    float h[4];
    h[0] = gelu_exact(a4.x + di2 * x4.x + bias[c0 + 0]);
    h[1] = gelu_exact(a4.y + di2 * x4.y + bias[c0 + 1]);
    h[2] = gelu_exact(a4.z + di2 * x4.z + bias[c0 + 2]);
    h[3] = gelu_exact(a4.w + di2 * x4.w + bias[c0 + 3]);

    float mean = wave_sum32(h[0] + h[1] + h[2] + h[3]) * (1.0f / 128.0f);
    float vs = 0.0f;
#pragma unroll
    for (int i = 0; i < 4; ++i) { float d = h[i] - mean; vs += d * d; }
    float var  = wave_sum32(vs) * (1.0f / 128.0f);
    float rstd = rsqrtf(var + LN_EPS);

    float* dst = out + (size_t)n * 128 + c0;
#pragma unroll
    for (int i = 0; i < 4; ++i)
        dst[i] = (h[i] - mean) * rstd * g[c0 + i] + bta[c0 + i];
}

// ---------------------------------------------------------------------------
// Edge update: ef = LN(ef + relu([o_r||o_c||ef] @ eu_w1 + b1) @ eu_w2 + b2)
// then ew = sigmoid(ef @ ew_w + ew_b), deg accumulate.  64 edges per block.
//   s_in  [ 64][328] bf16 @ 0       (41984)
//   s_w1t [128][328] bf16 @ 41984   (83968)   (= eu_w1^T)
//   s_h   [ 64][136] bf16 @ 125952  (17408)
//   s_w2t [ 64][136] bf16 @ 143360  (17408)   (= eu_w2^T)
//   s_efo [ 64][ 68] f32  @ 160768  (17408)
//   s_ef  [ 64][ 68] f32  @ 178176  (17408)
//   s_row [64] @ 195584, s_col @ 195840 ; total 196096 B
// ---------------------------------------------------------------------------
__global__ void edge_update_kernel(
    const float* __restrict__ o1, const int* __restrict__ ei,
    const float* __restrict__ w1, const float* __restrict__ b1,
    const float* __restrict__ w2, const float* __restrict__ b2,
    const float* __restrict__ lng, const float* __restrict__ lnb,
    const float* __restrict__ eww, const float* __restrict__ ewb,
    float* __restrict__ ef, float* __restrict__ ew,
    float* __restrict__ deg, int nE)
{
    extern __shared__ char smem[];
    __bf16* s_in  = (__bf16*)(smem);
    __bf16* s_w1t = (__bf16*)(smem + 41984);
    __bf16* s_h   = (__bf16*)(smem + 125952);
    __bf16* s_w2t = (__bf16*)(smem + 143360);
    float*  s_efo = (float*)(smem + 160768);
    float*  s_ef  = (float*)(smem + 178176);
    int*    s_row = (int*)(smem + 195584);
    int*    s_col = (int*)(smem + 195840);

    const int tid = threadIdx.x;
    const int e0  = blockIdx.x * 64;

    for (int i = tid; i < 64; i += 256) {
        int e = e0 + i; if (e > nE - 1) e = nE - 1;
        s_row[i] = ei[e];
        s_col[i] = ei[nE + e];
    }
    // old ef rows (f32, for residual)
    for (int idx = tid; idx < 64 * 16; idx += 256) {
        int le = idx >> 4, c4 = (idx & 15) << 2;
        int e = e0 + le; if (e > nE - 1) e = nE - 1;
        *(float4*)(s_efo + (size_t)le * 68 + c4) =
            *(const float4*)(ef + (size_t)e * 64 + c4);
    }
    __syncthreads();

    // concat [o1[row] || o1[col] || ef] -> bf16 LDS
    for (int idx = tid; idx < 64 * 80; idx += 256) {
        int le = idx / 80, c4 = (idx - le * 80) << 2;   // 0..316
        int e = e0 + le; if (e > nE - 1) e = nE - 1;
        float4 v;
        if (c4 < 128)      v = *(const float4*)(o1 + (size_t)s_row[le] * 128 + c4);
        else if (c4 < 256) v = *(const float4*)(o1 + (size_t)s_col[le] * 128 + (c4 - 128));
        else               v = *(const float4*)(ef + (size_t)e * 64 + (c4 - 256));
        *(v4bf*)(s_in + (size_t)le * 328 + c4) = pack4(v);
    }
    for (int idx = tid; idx < 320 * 128; idx += 256) {
        int k = idx >> 7, n = idx & 127;
        s_w1t[(size_t)n * 328 + k] = f2bf(w1[idx]);
    }
    for (int idx = tid; idx < 128 * 64; idx += 256) {
        int k = idx >> 6, n = idx & 63;
        s_w2t[(size_t)n * 136 + k] = f2bf(w2[idx]);
    }
    __syncthreads();

    const int wv = tid >> 5, lane = tid & 31;

    // GEMM1: [64,320] @ [320,128] -> relu -> s_h (bf16)
    // mt = wv&3 fixed per wave; nt = (wv>>2) + 2*i sweeps 4 N-tiles.
    {
        const int mt = wv & 3;
        v16bf afr[10];
#pragma unroll
        for (int kk = 0; kk < 10; ++kk)
            afr[kk] = load_frag_a(s_in, mt * 16, 328, kk * 32, lane);
        for (int i = 0; i < 4; ++i) {
            int nt = (wv >> 2) + 2 * i;
            v8f acc = {};
#pragma unroll
            for (int kk = 0; kk < 10; ++kk)
                acc = wmma_bf16(afr[kk],
                                load_frag_bt(s_w1t, nt * 16, 328, kk * 32, lane),
                                acc);
            int ccol  = nt * 16 + (lane & 15);
            float bi  = b1[ccol];
            int rbase = mt * 16 + ((lane >> 4) << 3);
#pragma unroll
            for (int r = 0; r < 8; ++r) {
                float v = acc[r] + bi;
                v = v > 0.0f ? v : 0.0f;
                s_h[(size_t)(rbase + r) * 136 + ccol] = f2bf(v);
            }
        }
    }
    __syncthreads();

    // GEMM2: [64,128] @ [128,64] + b2 + residual -> s_ef (f32)
    {
        const int mt = wv & 3;
        v16bf afr[4];
#pragma unroll
        for (int kk = 0; kk < 4; ++kk)
            afr[kk] = load_frag_a(s_h, mt * 16, 136, kk * 32, lane);
        for (int i = 0; i < 2; ++i) {
            int nt = (wv >> 2) + 2 * i;
            v8f acc = {};
#pragma unroll
            for (int kk = 0; kk < 4; ++kk)
                acc = wmma_bf16(afr[kk],
                                load_frag_bt(s_w2t, nt * 16, 136, kk * 32, lane),
                                acc);
            int ccol  = nt * 16 + (lane & 15);
            float bi  = b2[ccol];
            int rbase = mt * 16 + ((lane >> 4) << 3);
#pragma unroll
            for (int r = 0; r < 8; ++r) {
                int le = rbase + r;
                s_ef[(size_t)le * 68 + ccol] =
                    acc[r] + bi + s_efo[(size_t)le * 68 + ccol];
            }
        }
    }
    __syncthreads();

    // LayerNorm(64) per edge + new edge weight + degree.  One wave per edge.
    for (int le = wv; le < 64; le += 8) {
        float v0 = s_ef[(size_t)le * 68 + lane];
        float v1 = s_ef[(size_t)le * 68 + 32 + lane];
        float mean = wave_sum32(v0 + v1) * (1.0f / 64.0f);
        float d0 = v0 - mean, d1 = v1 - mean;
        float var  = wave_sum32(d0 * d0 + d1 * d1) * (1.0f / 64.0f);
        float rstd = rsqrtf(var + LN_EPS);
        float y0 = d0 * rstd * lng[lane]      + lnb[lane];
        float y1 = d1 * rstd * lng[32 + lane] + lnb[32 + lane];
        float dot = wave_sum32(y0 * eww[lane] + y1 * eww[32 + lane]) + ewb[0];
        float w = 1.0f / (1.0f + expf(-dot));
        int e = e0 + le;
        if (e < nE) {
            ef[(size_t)e * 64 + lane]      = y0;
            ef[(size_t)e * 64 + 32 + lane] = y1;
            if (lane == 0) {
                ew[e] = w;
                atomicAdd(&deg[s_col[le]], w);
            }
        }
    }
}

// ---------------------------------------------------------------------------
extern "C" void kernel_launch(void* const* d_in, const int* in_sizes, int n_in,
                              void* d_out, int out_size, void* d_ws, size_t ws_size,
                              hipStream_t stream) {
    const float* x      = (const float*)d_in[0];
    const int*   ei     = (const int*)d_in[1];
    const float* W1     = (const float*)d_in[2];
    const float* b1     = (const float*)d_in[3];
    const float* W2     = (const float*)d_in[4];
    const float* b2     = (const float*)d_in[5];
    const float* ln1_g  = (const float*)d_in[6];
    const float* ln1_b  = (const float*)d_in[7];
    const float* ln2_g  = (const float*)d_in[8];
    const float* ln2_b  = (const float*)d_in[9];
    const float* ei_w1  = (const float*)d_in[10];
    const float* ei_b1  = (const float*)d_in[11];
    const float* ei_w2  = (const float*)d_in[12];
    const float* ei_b2  = (const float*)d_in[13];
    const float* eu_w1  = (const float*)d_in[14];
    const float* eu_b1  = (const float*)d_in[15];
    const float* eu_w2  = (const float*)d_in[16];
    const float* eu_b2  = (const float*)d_in[17];
    const float* lne_g  = (const float*)d_in[18];
    const float* lne_b  = (const float*)d_in[19];
    const float* ew_w   = (const float*)d_in[20];
    const float* ew_b   = (const float*)d_in[21];

    const int N = in_sizes[0] / 128;
    const int E = in_sizes[1] / 2;

    // workspace carve (fp32): ef[E*64] ew[E] deg[N] dinv[N] xw[N*128] agg[N*128] out1[N*128]
    float* ws   = (float*)d_ws;
    float* ef   = ws;
    float* ewb  = ef   + (size_t)E * 64;
    float* deg  = ewb  + (size_t)E;
    float* dinv = deg  + (size_t)N;
    float* xw   = dinv + (size_t)N;
    float* agg  = xw   + (size_t)N * 128;
    float* out1 = agg  + (size_t)N * 128;

    const dim3 blk(256);
    const int zgrid = 2048;
    const int g_edge_init = (E + 127) / 128;
    const int g_node_gemm = (N + 127) / 128;
    const int g_scatter   = (int)(((long long)E * 32 + 255) / 256);
    const int g_nodefin   = (int)(((long long)N * 32 + 255) / 256);
    const int g_edge_upd  = (E + 63) / 64;
    const int g_dinv      = (N + 255) / 256;

    const size_t smem_einit = 164864;
    const size_t smem_gemm  = 69632;
    const size_t smem_eupd  = 196096;

    // ---- layer 1 ----
    zero_kernel<<<zgrid, blk, 0, stream>>>(deg, (long long)N);
    zero_kernel<<<zgrid, blk, 0, stream>>>(agg, (long long)N * 128);
    edge_init_kernel<<<g_edge_init, blk, smem_einit, stream>>>(
        x, ei, ei_w1, ei_b1, ei_w2, ei_b2, ew_w, ew_b, ef, ewb, deg, E);
    dinv_kernel<<<g_dinv, blk, 0, stream>>>(deg, dinv, N);
    gemm_node_kernel<<<g_node_gemm, blk, smem_gemm, stream>>>(x, W1, xw, N);
    scatter_kernel<<<g_scatter, blk, 0, stream>>>(ei, ewb, dinv, xw, agg, E);
    node_finalize_kernel<<<g_nodefin, blk, 0, stream>>>(
        agg, xw, dinv, b1, ln1_g, ln1_b, out1, N);

    // ---- edge update ----
    zero_kernel<<<zgrid, blk, 0, stream>>>(deg, (long long)N);
    zero_kernel<<<zgrid, blk, 0, stream>>>(agg, (long long)N * 128);
    edge_update_kernel<<<g_edge_upd, blk, smem_eupd, stream>>>(
        out1, ei, eu_w1, eu_b1, eu_w2, eu_b2, lne_g, lne_b, ew_w, ew_b,
        ef, ewb, deg, E);
    dinv_kernel<<<g_dinv, blk, 0, stream>>>(deg, dinv, N);

    // ---- layer 2 ----
    gemm_node_kernel<<<g_node_gemm, blk, smem_gemm, stream>>>(out1, W2, xw, N);
    scatter_kernel<<<g_scatter, blk, 0, stream>>>(ei, ewb, dinv, xw, agg, E);
    node_finalize_kernel<<<g_nodefin, blk, 0, stream>>>(
        agg, xw, dinv, b2, ln2_g, ln2_b, (float*)d_out, N);
}
